// Decoder_81252191305822
// MI455X (gfx1250) — matrix operand order repaired and checked by reading, not compiled, not run on hardware
//
#include <hip/hip_runtime.h>

// ---------------------------------------------------------------------------
// Tacotron2-style decoder for MI455X (gfx1250, wave32, WMMA).
//
// 200 sequential steps, ~1.2 GFLOP/step dominated by two LSTM-gate GEMMs with
// M=32. All recurrent weights (~35 MB as bf16) fit in the 192 MB L2, so the
// scan is L2-bandwidth / launch-latency bound, not HBM bound.
// => bf16 weights + v_wmma_f32_16x16x32_bf16 (f32 accumulate), and maximal
// per-step fusion: 6 kernels/step (gates, lstm+cvt, attention(+pq,+ctx_bf),
// gates, lstm+cvt, out/stop).
// ---------------------------------------------------------------------------

#define BB    32
#define TIN   256
#define TMEL  400
#define MELN  80
#define RR    2
#define ENC   512
#define PREN  256
#define QD    1024
#define DD    1024
#define AD    128
#define NF    32
#define KS    31
#define NSTEP 200
#define NGATE 4096

typedef __attribute__((ext_vector_type(16))) __bf16          v16bf;
typedef __attribute__((ext_vector_type(8)))  float           v8f;
typedef __attribute__((ext_vector_type(16))) unsigned short  v16u;
typedef __attribute__((ext_vector_type(8)))  unsigned short  v8u;

__device__ __forceinline__ unsigned short f2bf(float f) {
  unsigned u = __builtin_bit_cast(unsigned, f);
  u += 0x7FFFu + ((u >> 16) & 1u);          // round-to-nearest-even
  return (unsigned short)(u >> 16);
}
__device__ __forceinline__ float sigmoidf(float x) { return 1.0f / (1.0f + __expf(-x)); }

// ---------------------------------------------------------------------------
// WMMA fragment loads (bf16 stored as ushort).
// A (16x32, 16-bit) per-lane layout: lane L: M = L%16, h = L/16;
//   element j -> K = kb + 16h + j.
// B (32x16, 16-bit): lane L: N = L%16, h = L/16; element j -> K = kb + 16h + j.
// ---------------------------------------------------------------------------
__device__ __forceinline__ v16bf load_frag_a(const unsigned short* X, int ldx,
                                             int row, int kb, int h) {
  const unsigned short* p = X + (size_t)row * ldx + kb + 16 * h;
  v8u lo = *(const v8u*)p;
  v8u hi = *(const v8u*)(p + 8);
  v16u u;
#pragma unroll
  for (int j = 0; j < 8; ++j) { u[j] = lo[j]; u[8 + j] = hi[j]; }
  return __builtin_bit_cast(v16bf, u);
}
__device__ __forceinline__ v16bf load_frag_b(const unsigned short* wrow, int kb, int h) {
  return __builtin_bit_cast(v16bf, *(const v16u*)(wrow + kb + 16 * h));
}

__device__ __forceinline__ void gemm_seg(const unsigned short* __restrict__ X, int K,
                                         const unsigned short* __restrict__ wrow,
                                         int nl, int h, v8f& acc0, v8f& acc1) {
  for (int kb = 0; kb < K; kb += 32) {
    __builtin_prefetch(wrow + kb + 128, 0, 1);   // global_prefetch of B stream
    v16bf a0 = load_frag_a(X, K, nl, kb, h);        // rows M=0..15
    v16bf a1 = load_frag_a(X, K, nl + 16, kb, h);   // rows M=16..31
    v16bf bm = load_frag_b(wrow, kb, h);
    acc0 = __builtin_amdgcn_wmma_f32_16x16x32_bf16(false, a0, false, bm,
                                                   (short)0, acc0, false, false);
    acc1 = __builtin_amdgcn_wmma_f32_16x16x32_bf16(false, a1, false, bm,
                                                   (short)0, acc1, false, false);
  }
}

// gates[32][4096] = [X0|X1] @ W01^T + X2 @ W2^T + bih + bhh
// (segments 0,1 share W01 with column offsets 0 and K0; ld01 = K0+K1)
__global__ void __launch_bounds__(256) gates_wmma_kernel(
    const unsigned short* __restrict__ X0, int K0,
    const unsigned short* __restrict__ X1, int K1,
    const unsigned short* __restrict__ X2, int K2,
    const unsigned short* __restrict__ W01, int ld01,
    const unsigned short* __restrict__ W2,  int ld2,
    const float* __restrict__ bih, const float* __restrict__ bhh,
    float* __restrict__ gates) {
  int lane = threadIdx.x & 31;
  int wave = threadIdx.x >> 5;
  int h  = lane >> 4;
  int nl = lane & 15;
  int ncol = blockIdx.x * 128 + wave * 16 + nl;
  v8f acc0 = {};
  v8f acc1 = {};
  const unsigned short* w0 = W01 + (size_t)ncol * ld01;
  const unsigned short* w1 = w0 + K0;
  const unsigned short* w2 = W2 + (size_t)ncol * ld2;
  gemm_seg(X0, K0, w0, nl, h, acc0, acc1);
  gemm_seg(X1, K1, w1, nl, h, acc0, acc1);
  gemm_seg(X2, K2, w2, nl, h, acc0, acc1);
  float bias = bih[ncol] + bhh[ncol];
  // C/D layout: VGPR r, lane L: N = L%16, M = r + 8*(L/16)
#pragma unroll
  for (int r = 0; r < 8; ++r) {
    gates[(size_t)(r + 8 * h) * NGATE + ncol]      = acc0[r] + bias;
    gates[(size_t)(16 + r + 8 * h) * NGATE + ncol] = acc1[r] + bias;
  }
}

// ---------------------------------------------------------------------------
// LSTM cell update fused with bf16 conversion of the new hidden state.
__global__ void __launch_bounds__(256) lstm_update_kernel(
    const float* __restrict__ gates, float* __restrict__ h, float* __restrict__ c,
    unsigned short* __restrict__ h_bf) {
  int idx = blockIdx.x * 256 + threadIdx.x;   // 0..32767
  int b = idx >> 10;
  int n = idx & 1023;
  const float* g = gates + (size_t)b * NGATE;
  float gi = g[n], gf = g[n + QD], gg = g[n + 2 * QD], go = g[n + 3 * QD];
  float cn = sigmoidf(gf) * c[idx] + sigmoidf(gi) * tanhf(gg);
  float hn = sigmoidf(go) * tanhf(cn);
  c[idx] = cn;
  h[idx] = hn;
  h_bf[idx] = f2bf(hn);
}

// Fused location-sensitive attention:
//   pq = h_a @ wq^T, loc-conv + energies + softmax + context (+ bf16 ctx).
// One block per batch element, 256 threads (= Tin).
__global__ void __launch_bounds__(256) attention_kernel(
    const float* __restrict__ h_a, const float* __restrict__ wq,
    float* __restrict__ aw, float* __restrict__ awc,
    const float* __restrict__ proc,
    const float* __restrict__ conv_w, const float* __restrict__ dense_w,
    const float* __restrict__ v_w, const float* __restrict__ v_b,
    const unsigned char* __restrict__ mask,   // jnp bool -> 1 byte/elem
    const float* __restrict__ inputs,
    float* __restrict__ ctx, unsigned short* __restrict__ ctx_bf,
    float* __restrict__ out_align, int step) {
  __shared__ float s_cat[2][TIN + KS - 1];
  __shared__ float s_ld[AD * NF];
  __shared__ float s_cw[NF * 2 * KS];
  __shared__ float s_vw[AD];
  __shared__ float s_pq[AD];
  __shared__ float s_align[TIN];
  __shared__ float s_red[TIN];
  int b = blockIdx.x;
  int t = threadIdx.x;

  // ---- pq = h_a[b] @ wq^T : 2 threads per output, 512-elem partial dots ----
  {
    int a = t >> 1, half = t & 1;
    const float* x = h_a + (size_t)b * QD + half * (QD / 2);
    const float* w = wq + (size_t)a * QD + half * (QD / 2);
    float s = 0.f;
    for (int k = 0; k < QD / 2; ++k) s += x[k] * w[k];
    s_red[t] = s;
  }
  // ---- stage shared operands ----
  s_cat[0][15 + t] = aw[b * TIN + t];
  s_cat[1][15 + t] = awc[b * TIN + t];
  if (t < 15) {
    s_cat[0][t] = 0.f; s_cat[1][t] = 0.f;
    s_cat[0][15 + TIN + t] = 0.f; s_cat[1][15 + TIN + t] = 0.f;
  }
  for (int i = t; i < AD * NF; i += 256) s_ld[i] = dense_w[i];
  for (int i = t; i < NF * 2 * KS; i += 256) s_cw[i] = conv_w[i];
  if (t < AD) s_vw[t] = v_w[t];
  __syncthreads();
  if (t < AD) s_pq[t] = s_red[2 * t] + s_red[2 * t + 1];
  __syncthreads();

  // ---- location conv (cross-correlation, pad 15), per output position t ----
  float loc[NF];
  for (int c = 0; c < NF; ++c) {
    float a = 0.f;
    const float* wc = &s_cw[c * 2 * KS];
#pragma unroll
    for (int k = 0; k < KS; ++k) {
      a += s_cat[0][t + k] * wc[k];
      a += s_cat[1][t + k] * wc[KS + k];
    }
    loc[c] = a;
  }
  // ---- energy e[t] = vb + sum_a vw[a]*tanh(pq[a] + ploc[t,a] + proc[t,a]) ----
  const float* procb = proc + ((size_t)b * TIN + t) * AD;
  float e = v_b[0];
  for (int a = 0; a < AD; ++a) {
    float s = s_pq[a] + procb[a];
#pragma unroll
    for (int c = 0; c < NF; ++c) s += s_ld[a * NF + c] * loc[c];
    e += s_vw[a] * tanhf(s);
  }
  if (!mask[b * TIN + t]) e = -1.0e9f;

  // ---- softmax over 256 ----
  __syncthreads();
  s_red[t] = e; __syncthreads();
  for (int s = 128; s > 0; s >>= 1) { if (t < s) s_red[t] = fmaxf(s_red[t], s_red[t + s]); __syncthreads(); }
  float mx = s_red[0]; __syncthreads();
  float ex = __expf(e - mx);
  s_red[t] = ex; __syncthreads();
  for (int s = 128; s > 0; s >>= 1) { if (t < s) s_red[t] += s_red[t + s]; __syncthreads(); }
  float al = ex / s_red[0];
  s_align[t] = al;
  aw[b * TIN + t] = al;
  awc[b * TIN + t] += al;
  out_align[((size_t)b * NSTEP + step) * TIN + t] = al;
  __syncthreads();

  // ---- ctx[b][d] = sum_t align[t] * inputs[b][t][d]  (+ bf16 copy) ----
  const float* inpb = inputs + (size_t)b * TIN * ENC;
  for (int d = t; d < ENC; d += 256) {
    float s = 0.f;
    for (int tt = 0; tt < TIN; ++tt) s += s_align[tt] * inpb[(size_t)tt * ENC + d];
    ctx[b * ENC + d] = s;
    ctx_bf[b * ENC + d] = f2bf(s);
  }
}

// out = [h_d|ctx] @ wp^T + bp (written transposed into mel output);
// stop = [h_d|out] @ ws^T + bs
__global__ void __launch_bounds__(256) outstop_kernel(
    const float* __restrict__ h_d, const float* __restrict__ ctx,
    const float* __restrict__ wp, const float* __restrict__ bp,
    const float* __restrict__ wsw, const float* __restrict__ bs,
    float* __restrict__ out_mel, float* __restrict__ out_stop, int step) {
  __shared__ float s_out[RR * MELN];
  __shared__ float s_red[256];
  int b = blockIdx.x, t = threadIdx.x;
  const float* hd = h_d + (size_t)b * DD;
  const float* cx = ctx + (size_t)b * ENC;
  if (t < RR * MELN) {
    const float* w = wp + (size_t)t * (DD + ENC);
    float s = bp[t];
    for (int k = 0; k < DD; ++k)  s += hd[k] * w[k];
    for (int k = 0; k < ENC; ++k) s += cx[k] * w[DD + k];
    s_out[t] = s;
    int frame = RR * step + t / MELN;
    int m = t % MELN;
    out_mel[((size_t)b * MELN + m) * TMEL + frame] = s;
  }
  __syncthreads();
  float p = 0.f;
  for (int i = t; i < DD + RR * MELN; i += 256) {
    float x = (i < DD) ? hd[i] : s_out[i - DD];
    p += x * wsw[i];
  }
  s_red[t] = p; __syncthreads();
  for (int s = 128; s > 0; s >>= 1) { if (t < s) s_red[t] += s_red[t + s]; __syncthreads(); }
  if (t == 0) out_stop[(size_t)b * NSTEP + step] = s_red[0] + bs[0];
}

// ---------------------------------------------------------------------------
// One-time prep kernels
// ---------------------------------------------------------------------------
__global__ void build_mem_kernel(const float* __restrict__ memories, float* __restrict__ mem) {
  int idx = blockIdx.x * blockDim.x + threadIdx.x;
  int total = (NSTEP + 1) * BB * MELN;
  if (idx >= total) return;
  int t = idx / (BB * MELN);
  int rem = idx % (BB * MELN);
  int b = rem / MELN;
  int m = rem % MELN;
  float v = 0.f;
  if (t > 0) v = memories[((size_t)b * TMEL + (RR * (t - 1) + 1)) * MELN + m];
  mem[idx] = v;
}
__global__ void __launch_bounds__(256) prenet1_kernel(
    const float* __restrict__ mem, const float* __restrict__ w1, float* __restrict__ h1) {
  int row = blockIdx.x, j = threadIdx.x;
  const float* x = mem + (size_t)row * MELN;
  const float* w = w1 + (size_t)j * MELN;
  float s = 0.f;
  for (int k = 0; k < MELN; ++k) s += x[k] * w[k];
  h1[(size_t)row * PREN + j] = fmaxf(s, 0.f);
}
__global__ void __launch_bounds__(256) prenet2_kernel(
    const float* __restrict__ h1, const float* __restrict__ w2,
    float* __restrict__ pre, unsigned short* __restrict__ pre_bf) {
  int row = blockIdx.x, j = threadIdx.x;
  const float* x = h1 + (size_t)row * PREN;
  const float* w = w2 + (size_t)j * PREN;
  float s = 0.f;
  for (int k = 0; k < PREN; ++k) s += x[k] * w[k];
  float r = fmaxf(s, 0.f);
  pre[(size_t)row * PREN + j] = r;
  pre_bf[(size_t)row * PREN + j] = f2bf(r);
}
__global__ void __launch_bounds__(128) proc_kernel(
    const float* __restrict__ inputs, const float* __restrict__ win, float* __restrict__ proc) {
  int row = blockIdx.x, a = threadIdx.x;
  const float* x = inputs + (size_t)row * ENC;
  const float* w = win + (size_t)a * ENC;
  float s = 0.f;
  for (int k = 0; k < ENC; ++k) s += x[k] * w[k];
  proc[(size_t)row * AD + a] = s;
}
__global__ void cvt_bf16_kernel(const float* __restrict__ src, unsigned short* __restrict__ dst, int n) {
  int i = blockIdx.x * blockDim.x + threadIdx.x;
  if (i < n) dst[i] = f2bf(src[i]);
}
__global__ void zero_kernel(unsigned int* __restrict__ p, int n) {
  int i = blockIdx.x * blockDim.x + threadIdx.x;
  if (i < n) p[i] = 0u;
}

// ---------------------------------------------------------------------------
extern "C" void kernel_launch(void* const* d_in, const int* in_sizes, int n_in,
                              void* d_out, int out_size, void* d_ws, size_t ws_size,
                              hipStream_t stream) {
  (void)in_sizes; (void)n_in; (void)out_size; (void)ws_size;
  const float* inputs       = (const float*)d_in[0];
  const float* memories     = (const float*)d_in[1];
  const unsigned char* mask = (const unsigned char*)d_in[2];
  const float* prenet_w1    = (const float*)d_in[3];
  const float* prenet_w2    = (const float*)d_in[4];
  const float* wih_a = (const float*)d_in[5];
  const float* whh_a = (const float*)d_in[6];
  const float* bih_a = (const float*)d_in[7];
  const float* bhh_a = (const float*)d_in[8];
  const float* wq    = (const float*)d_in[9];
  const float* win   = (const float*)d_in[10];
  const float* v_w   = (const float*)d_in[11];
  const float* v_b   = (const float*)d_in[12];
  const float* conv_w  = (const float*)d_in[13];
  const float* dense_w = (const float*)d_in[14];
  const float* wih_d = (const float*)d_in[15];
  const float* whh_d = (const float*)d_in[16];
  const float* bih_d = (const float*)d_in[17];
  const float* bhh_d = (const float*)d_in[18];
  const float* wp  = (const float*)d_in[19];
  const float* bp  = (const float*)d_in[20];
  const float* wsw = (const float*)d_in[21];
  const float* bs  = (const float*)d_in[22];

  float* out_mel   = (float*)d_out;                                  // (B,80,400)
  float* out_align = out_mel + (size_t)BB * MELN * TMEL;             // (B,200,256)
  float* out_stop  = out_align + (size_t)BB * NSTEP * TIN;           // (B,200,1)

  char* wsp = (char*)d_ws;
  size_t off = 0;
  auto alloc = [&](size_t bytes) -> void* {
    void* r = wsp + off;
    off = (off + bytes + 255) & ~(size_t)255;
    return r;
  };
  unsigned short* wih_a_bf = (unsigned short*)alloc((size_t)NGATE * 768  * 2);
  unsigned short* whh_a_bf = (unsigned short*)alloc((size_t)NGATE * 1024 * 2);
  unsigned short* wih_d_bf = (unsigned short*)alloc((size_t)NGATE * 1536 * 2);
  unsigned short* whh_d_bf = (unsigned short*)alloc((size_t)NGATE * 1024 * 2);
  float* memb  = (float*)alloc((size_t)(NSTEP + 1) * BB * MELN * 4);
  float* ph1   = (float*)alloc((size_t)(NSTEP + 1) * BB * PREN * 4);
  float* pre   = (float*)alloc((size_t)(NSTEP + 1) * BB * PREN * 4);
  unsigned short* pre_bf = (unsigned short*)alloc((size_t)(NSTEP + 1) * BB * PREN * 2);
  float* proc  = (float*)alloc((size_t)BB * TIN * AD * 4);
  float* gates = (float*)alloc((size_t)BB * NGATE * 4);
  size_t state_begin = off;
  float* h_a = (float*)alloc((size_t)BB * QD * 4);
  float* c_a = (float*)alloc((size_t)BB * QD * 4);
  float* h_d = (float*)alloc((size_t)BB * DD * 4);
  float* c_d = (float*)alloc((size_t)BB * DD * 4);
  float* ctx = (float*)alloc((size_t)BB * ENC * 4);
  float* aw  = (float*)alloc((size_t)BB * TIN * 4);
  float* awc = (float*)alloc((size_t)BB * TIN * 4);
  unsigned short* h_a_bf = (unsigned short*)alloc((size_t)BB * QD * 2);
  unsigned short* h_d_bf = (unsigned short*)alloc((size_t)BB * DD * 2);
  unsigned short* ctx_bf = (unsigned short*)alloc((size_t)BB * ENC * 2);
  size_t state_bytes = off - state_begin;

  // -------- per-call prep (deterministic) --------
  {
    int n;
    n = NGATE * 768;  cvt_bf16_kernel<<<(n + 255) / 256, 256, 0, stream>>>(wih_a, wih_a_bf, n);
    n = NGATE * 1024; cvt_bf16_kernel<<<(n + 255) / 256, 256, 0, stream>>>(whh_a, whh_a_bf, n);
    n = NGATE * 1536; cvt_bf16_kernel<<<(n + 255) / 256, 256, 0, stream>>>(wih_d, wih_d_bf, n);
    n = NGATE * 1024; cvt_bf16_kernel<<<(n + 255) / 256, 256, 0, stream>>>(whh_d, whh_d_bf, n);
    n = (int)(state_bytes / 4);
    zero_kernel<<<(n + 255) / 256, 256, 0, stream>>>((unsigned int*)h_a, n);
    n = (NSTEP + 1) * BB * MELN;
    build_mem_kernel<<<(n + 255) / 256, 256, 0, stream>>>(memories, memb);
    prenet1_kernel<<<(NSTEP + 1) * BB, 256, 0, stream>>>(memb, prenet_w1, ph1);
    prenet2_kernel<<<(NSTEP + 1) * BB, 256, 0, stream>>>(ph1, prenet_w2, pre, pre_bf);
    proc_kernel<<<BB * TIN, 128, 0, stream>>>(inputs, win, proc);
  }

  // -------- sequential decoder scan: 6 kernels per step --------
  for (int t = 0; t < NSTEP; ++t) {
    // attention LSTM: gates = [m_t|ctx] @ wih_a^T + h_a @ whh_a^T + biases
    gates_wmma_kernel<<<32, 256, 0, stream>>>(
        pre_bf + (size_t)t * BB * PREN, PREN, ctx_bf, ENC, h_a_bf, QD,
        wih_a_bf, PREN + ENC, whh_a_bf, QD, bih_a, bhh_a, gates);
    lstm_update_kernel<<<BB * QD / 256, 256, 0, stream>>>(gates, h_a, c_a, h_a_bf);
    attention_kernel<<<BB, 256, 0, stream>>>(h_a, wq, aw, awc, proc, conv_w, dense_w,
                                             v_w, v_b, mask, inputs, ctx, ctx_bf,
                                             out_align, t);
    // decoder LSTM: gates = [h_a|ctx] @ wih_d^T + h_d @ whh_d^T + biases
    gates_wmma_kernel<<<32, 256, 0, stream>>>(
        h_a_bf, QD, ctx_bf, ENC, h_d_bf, DD,
        wih_d_bf, QD + ENC, whh_d_bf, DD, bih_d, bhh_d, gates);
    lstm_update_kernel<<<BB * DD / 256, 256, 0, stream>>>(gates, h_d, c_d, h_d_bf);
    outstop_kernel<<<BB, 256, 0, stream>>>(h_d, ctx, wp, bp, wsw, bs, out_mel, out_stop, t);
  }
}
